// WindowAttention_65034394796652
// MI455X (gfx1250) — matrix operand order repaired and checked by reading, not compiled, not run on hardware
//
#include <hip/hip_runtime.h>
#include <hip/hip_bf16.h>
#include <math.h>

// ---------------------------------------------------------------------------
// Shifted-window attention (Swin-style) for MI455X / gfx1250.
// All three matmul stages run on v_wmma_f32_16x16x32_bf16 (fp32 accumulate),
// softmax/masks in fp32.  Intermediates stored bf16 to halve HBM traffic.
// ---------------------------------------------------------------------------

typedef __bf16 bf16_t;
typedef __attribute__((ext_vector_type(16))) __bf16 v16bf;
typedef __attribute__((ext_vector_type(8)))  float  v8f;

#define B_      8
#define HH_     96
#define WW_     96
#define DIM_    384
#define HEADS_  12
#define HD_     32
#define WS_     12
#define DISP_   6
#define NW_     64
#define WSQ_    144
#define INNER_  384
#define NQKV_   1152
#define MTOT_   (B_*HH_*WW_)     /* 73728 */

// round-to-nearest-even fp32 -> bf16 (bit pattern as ushort)
__device__ __forceinline__ unsigned short f2bfu(float f) {
  union { float f; unsigned u; } x; x.f = f;
  unsigned r = x.u + 0x7FFFu + ((x.u >> 16) & 1u);
  return (unsigned short)(r >> 16);
}

__device__ __forceinline__ v8f wmma_bf16(v16bf a, v16bf b, v8f c) {
  return __builtin_amdgcn_wmma_f32_16x16x32_bf16(
      /*neg_a=*/false, a, /*neg_b=*/false, b,
      /*c_mod=*/(short)0, c, /*reuse_a=*/false, /*reuse_b=*/false);
}

// A fragment: 16x32 bf16, rows M in [base .. base+15], ld = row stride (elems).
// lane L: M=L&15, half=L>>4; elems e -> K = half*16 + e (16 contiguous).
__device__ __forceinline__ v16bf ld_frag_a(const unsigned short* base, int ld) {
  const int lane = threadIdx.x & 31;
  const int m = lane & 15, h = lane >> 4;
  const unsigned short* p = base + m * ld + h * 16;
  union { v16bf v; uint4 q[2]; } f;
  f.q[0] = *(const uint4*)(p);
  f.q[1] = *(const uint4*)(p + 8);
  return f.v;
}

// B fragment: 32x16 bf16 (KxN) read from N-major storage [n][k], ld = k stride.
// lane L: N=L&15, half=L>>4; elems e -> K = half*16 + e (16 contiguous).
__device__ __forceinline__ v16bf ld_frag_b(const unsigned short* base, int ld) {
  const int lane = threadIdx.x & 31;
  const int n = lane & 15, h = lane >> 4;
  const unsigned short* p = base + n * ld + h * 16;
  union { v16bf v; uint4 q[2]; } f;
  f.q[0] = *(const uint4*)(p);
  f.q[1] = *(const uint4*)(p + 8);
  return f.v;
}

// ---------------------------------------------------------------------------
// Kernel 0: convert weights to bf16, transposed to N-major [n][k].
// ---------------------------------------------------------------------------
__global__ void k_convert(const float* __restrict__ wqkv,
                          const float* __restrict__ wout,
                          unsigned short* __restrict__ wqkvT,
                          unsigned short* __restrict__ woutT) {
  int t = blockIdx.x * blockDim.x + threadIdx.x;
  if (t < NQKV_ * DIM_) {
    int n = t / DIM_, k = t - n * DIM_;
    wqkvT[t] = f2bfu(wqkv[k * NQKV_ + n]);
  }
  int t2 = t - NQKV_ * DIM_;
  if (t2 >= 0 && t2 < INNER_ * DIM_) {
    int n = t2 / DIM_, k = t2 - n * DIM_;
    woutT[t2] = f2bfu(wout[k * INNER_ + n]);
  }
}

// ---------------------------------------------------------------------------
// Kernel 1: shifted QKV projection. C[73728,1152] = Xshift @ Wqkv.
// 64x64 tile per workgroup, 4 waves, each wave a 32x32 quadrant (2x2 WMMA).
// Output scattered as bf16 directly into window layout [B,HEADS,NW,144,32].
// ---------------------------------------------------------------------------
__global__ __launch_bounds__(128) void k_qkv(
    const float* __restrict__ x, const unsigned short* __restrict__ wqkvT,
    unsigned short* __restrict__ Qw, unsigned short* __restrict__ Kw,
    unsigned short* __restrict__ Vw) {
  __shared__ unsigned short Xt[64 * 32];   // A tile, row-major [m][k]
  __shared__ unsigned short Wt[64 * 32];   // B tile, N-major  [n][k]
  const int m0 = blockIdx.y * 64;
  const int n0 = blockIdx.x * 64;
  const int tid = threadIdx.x;
  const int wv = tid >> 5, lane = tid & 31;

  v8f acc[2][2];
#pragma unroll
  for (int a = 0; a < 2; ++a)
#pragma unroll
    for (int b = 0; b < 2; ++b)
      acc[a][b] = (v8f){0.f,0.f,0.f,0.f,0.f,0.f,0.f,0.f};

  const int r = tid >> 1, half = tid & 1;   // 2 threads per tile row
  const unsigned m   = (unsigned)(m0 + r);
  const unsigned bb  = m / 9216u, rem = m % 9216u;
  const unsigned ii  = rem / 96u, jj = rem % 96u;
  const unsigned si  = (ii + DISP_) % 96u, sj = (jj + DISP_) % 96u;
  const float* xrow  = x + ((bb * 96u + si) * 96u + sj) * 384u;

  for (int kk = 0; kk < DIM_; kk += 32) {
    // stage A: fp32 -> bf16
    const float4* src = (const float4*)(xrow + kk + half * 16);
    unsigned short* dst = &Xt[r * 32 + half * 16];
#pragma unroll
    for (int q4 = 0; q4 < 4; ++q4) {
      float4 f = src[q4];
      dst[q4 * 4 + 0] = f2bfu(f.x); dst[q4 * 4 + 1] = f2bfu(f.y);
      dst[q4 * 4 + 2] = f2bfu(f.z); dst[q4 * 4 + 3] = f2bfu(f.w);
    }
    // stage B: already bf16, contiguous copy
    {
      const uint4* ws_ = (const uint4*)(wqkvT + (unsigned)(n0 + r) * 384u + kk + half * 16);
      uint4* wd = (uint4*)&Wt[r * 32 + half * 16];
      wd[0] = ws_[0]; wd[1] = ws_[1];
    }
    __syncthreads();
    const int msA = (wv >> 1) * 32;
    const int nsB = (wv & 1) * 32;
    v16bf af0 = ld_frag_a(&Xt[(msA +  0) * 32], 32);
    v16bf af1 = ld_frag_a(&Xt[(msA + 16) * 32], 32);
    v16bf bf0 = ld_frag_b(&Wt[(nsB +  0) * 32], 32);
    v16bf bf1 = ld_frag_b(&Wt[(nsB + 16) * 32], 32);
    acc[0][0] = wmma_bf16(af0, bf0, acc[0][0]);
    acc[0][1] = wmma_bf16(af0, bf1, acc[0][1]);
    acc[1][0] = wmma_bf16(af1, bf0, acc[1][0]);
    acc[1][1] = wmma_bf16(af1, bf1, acc[1][1]);
    __syncthreads();
  }

  const int h = lane >> 4, ln = lane & 15;
#pragma unroll
  for (int a = 0; a < 2; ++a)
#pragma unroll
    for (int bq = 0; bq < 2; ++bq)
#pragma unroll
      for (int rr = 0; rr < 8; ++rr) {
        unsigned mm = (unsigned)(m0 + (wv >> 1) * 32 + a * 16 + h * 8 + rr);
        unsigned nn = (unsigned)(n0 + (wv & 1) * 32 + bq * 16 + ln);
        unsigned b2 = mm / 9216u, rm = mm % 9216u;
        unsigned i2 = rm / 96u, j2 = rm % 96u;
        unsigned wwin = (i2 / 12u) * 8u + (j2 / 12u);
        unsigned pp = (i2 % 12u) * 12u + (j2 % 12u);
        unsigned which = nn / 384u;
        unsigned head = (nn % 384u) / 32u;
        unsigned dd = nn % 32u;
        unsigned short* dstp = (which == 0u) ? Qw : ((which == 1u) ? Kw : Vw);
        dstp[(((b2 * 12u + head) * 64u + wwin) * 144u + pp) * 32u + dd]
            = f2bfu(acc[a][bq][rr]);
      }
}

// ---------------------------------------------------------------------------
// Kernel 2: attention. One workgroup (9 waves) per (b, head, window).
// Wave wv owns a 16-row strip of the 144x144 score matrix.
// ---------------------------------------------------------------------------
__global__ __launch_bounds__(288) void k_attn(
    const unsigned short* __restrict__ Qw, const unsigned short* __restrict__ Kw,
    const unsigned short* __restrict__ Vw, const int* __restrict__ pix_map,
    const float* __restrict__ pos_emb, const float* __restrict__ pix_emb,
    unsigned short* __restrict__ AO) {
  __shared__ float          sbuf[144 * 144];   // S fp32; reused as P bf16 (stride 288)
  __shared__ unsigned short qs[144 * 32];
  __shared__ unsigned short ks[144 * 32];
  __shared__ unsigned short vt[32 * 160];      // v transposed [d][key], key padded ->160
  __shared__ float          posb[529];
  __shared__ float          pmv[144];

  const int idx  = blockIdx.x;
  const int bb   = idx / (HEADS_ * NW_);
  const int head = (idx / NW_) % HEADS_;
  const int wwin = idx % NW_;
  const int wi = wwin >> 3, wj = wwin & 7;
  const int tid = threadIdx.x;
  const int wv = tid >> 5, lane = tid & 31;
  const unsigned base = (unsigned)((bb * 12 + head) * 64 + wwin) * (144u * 32u);

  { // stage q,k (contiguous 128-bit copies: 576 uint4 each, 288 threads)
    const uint4* gq = (const uint4*)(Qw + base);
    const uint4* gk = (const uint4*)(Kw + base);
    ((uint4*)qs)[tid]       = gq[tid];
    ((uint4*)qs)[tid + 288] = gq[tid + 288];
    ((uint4*)ks)[tid]       = gk[tid];
    ((uint4*)ks)[tid + 288] = gk[tid + 288];
  }
  { // stage v transposed with zero pad for keys 144..159
    const unsigned short* gv = Vw + base;
    for (int l = tid; l < 32 * 160; l += 288) {
      int d = l / 160, key = l - d * 160;
      vt[l] = (key < 144) ? gv[(unsigned)key * 32u + d] : (unsigned short)0;
    }
  }
  for (int l = tid; l < 529; l += 288) posb[l] = pos_emb[l];
  if (tid < 144) {
    int px = tid / 12, py = tid - px * 12;
    pmv[tid] = (float)pix_map[(bb * 96 + wi * 12 + px) * 96 + (wj * 12 + py)];
  }
  const float pe0 = pix_emb[0], pe1 = pix_emb[1];
  __syncthreads();

  // ---- S = (q k^T) * scale + pix + posbias + shift masks ----
  const float scale = 0.17677669529663687f;   // 32^-0.5
  const int h = lane >> 4, ln = lane & 15;
  v16bf aq = ld_frag_a(&qs[(wv * 16) * 32], 32);
  for (int ct = 0; ct < 9; ++ct) {
    v16bf bk = ld_frag_b(&ks[(ct * 16) * 32], 32);
    v8f s = (v8f){0.f,0.f,0.f,0.f,0.f,0.f,0.f,0.f};
    s = wmma_bf16(aq, bk, s);
    const int jcol = ct * 16 + ln;
    const int jx = jcol / 12, jy = jcol - jx * 12;
#pragma unroll
    for (int rr = 0; rr < 8; ++rr) {
      const int irow = wv * 16 + h * 8 + rr;
      const int ix = irow / 12, iy = irow - ix * 12;
      float val = s[rr] * scale;
      val += (pmv[irow] * pmv[jcol] != 0.f) ? pe1 : pe0;
      val += posb[(jx - ix + 11) * 23 + (jy - iy + 11)];
      if (wi == 7 && ((irow >= 72) != (jcol >= 72))) val = -__builtin_inff();
      if (wj == 7 && ((iy >= 6) != (jy >= 6)))       val = -__builtin_inff();
      sbuf[irow * 144 + jcol] = val;
    }
  }
  __syncthreads();

  // ---- softmax per row; write P bf16 in place (2c write < 4c read: safe) ----
  if (tid < 144) {
    float* srow = &sbuf[tid * 144];
    float mx = -__builtin_inff();
    for (int c = 0; c < 144; ++c) mx = fmaxf(mx, srow[c]);
    float sum = 0.f;
    for (int c = 0; c < 144; ++c) sum += __expf(srow[c] - mx);
    const float inv = 1.f / sum;
    unsigned short* prow = (unsigned short*)srow;
    for (int c = 0; c < 144; ++c) {
      float e = __expf(srow[c] - mx) * inv;
      prow[c] = f2bfu(e);
    }
    for (int c = 144; c < 160; ++c) prow[c] = 0;   // zero-pad K to 160
  }
  __syncthreads();

  // ---- O = P @ v  (K padded to 160 = 5 WMMA steps, 2 N-tiles) ----
  const unsigned short* Pb = (const unsigned short*)sbuf;  // row stride 288 bf16
  v8f o0 = (v8f){0.f,0.f,0.f,0.f,0.f,0.f,0.f,0.f};
  v8f o1 = o0;
#pragma unroll
  for (int kt = 0; kt < 5; ++kt) {
    v16bf ap = ld_frag_a(Pb + (wv * 16) * 288 + kt * 32, 288);
    v16bf bv0 = ld_frag_b(&vt[ 0 * 160 + kt * 32], 160);
    v16bf bv1 = ld_frag_b(&vt[16 * 160 + kt * 32], 160);
    o0 = wmma_bf16(ap, bv0, o0);
    o1 = wmma_bf16(ap, bv1, o1);
  }
#pragma unroll
  for (int nt = 0; nt < 2; ++nt) {
    const v8f& o = nt ? o1 : o0;
#pragma unroll
    for (int rr = 0; rr < 8; ++rr) {
      const int p = wv * 16 + h * 8 + rr;
      const int dd = nt * 16 + ln;
      const int gi = wi * 12 + p / 12, gj = wj * 12 + p % 12;
      AO[((unsigned)(bb * 96 + gi) * 96u + gj) * 384u + head * 32u + dd] = f2bfu(o[rr]);
    }
  }
}

// ---------------------------------------------------------------------------
// Kernel 3: output projection + bias + inverse cyclic roll.
// C[73728,384] = AO @ Wout + b; store at rolled (i+6, j+6) position.
// ---------------------------------------------------------------------------
__global__ __launch_bounds__(128) void k_out(
    const unsigned short* __restrict__ AO, const unsigned short* __restrict__ woutT,
    const float* __restrict__ bout, float* __restrict__ out) {
  __shared__ unsigned short At[64 * 32];
  __shared__ unsigned short Bt[64 * 32];
  const int m0 = blockIdx.y * 64;
  const int n0 = blockIdx.x * 64;
  const int tid = threadIdx.x;
  const int wv = tid >> 5, lane = tid & 31;

  v8f acc[2][2];
#pragma unroll
  for (int a = 0; a < 2; ++a)
#pragma unroll
    for (int b = 0; b < 2; ++b)
      acc[a][b] = (v8f){0.f,0.f,0.f,0.f,0.f,0.f,0.f,0.f};

  const int r = tid >> 1, half = tid & 1;
  for (int kk = 0; kk < DIM_; kk += 32) {
    {
      const uint4* as_ = (const uint4*)(AO + (unsigned)(m0 + r) * 384u + kk + half * 16);
      uint4* ad = (uint4*)&At[r * 32 + half * 16];
      ad[0] = as_[0]; ad[1] = as_[1];
      const uint4* bs_ = (const uint4*)(woutT + (unsigned)(n0 + r) * 384u + kk + half * 16);
      uint4* bd = (uint4*)&Bt[r * 32 + half * 16];
      bd[0] = bs_[0]; bd[1] = bs_[1];
    }
    __syncthreads();
    const int msA = (wv >> 1) * 32;
    const int nsB = (wv & 1) * 32;
    v16bf af0 = ld_frag_a(&At[(msA +  0) * 32], 32);
    v16bf af1 = ld_frag_a(&At[(msA + 16) * 32], 32);
    v16bf bf0 = ld_frag_b(&Bt[(nsB +  0) * 32], 32);
    v16bf bf1 = ld_frag_b(&Bt[(nsB + 16) * 32], 32);
    acc[0][0] = wmma_bf16(af0, bf0, acc[0][0]);
    acc[0][1] = wmma_bf16(af0, bf1, acc[0][1]);
    acc[1][0] = wmma_bf16(af1, bf0, acc[1][0]);
    acc[1][1] = wmma_bf16(af1, bf1, acc[1][1]);
    __syncthreads();
  }

  const int h = lane >> 4, ln = lane & 15;
#pragma unroll
  for (int a = 0; a < 2; ++a)
#pragma unroll
    for (int bq = 0; bq < 2; ++bq) {
      const unsigned nn = (unsigned)(n0 + (wv & 1) * 32 + bq * 16 + ln);
      const float bias = bout[nn];
#pragma unroll
      for (int rr = 0; rr < 8; ++rr) {
        unsigned mm = (unsigned)(m0 + (wv >> 1) * 32 + a * 16 + h * 8 + rr);
        unsigned b2 = mm / 9216u, rm = mm % 9216u;
        unsigned i2 = rm / 96u, j2 = rm % 96u;
        unsigned oi = (i2 + DISP_) % 96u, oj = (j2 + DISP_) % 96u;  // inverse roll
        out[((b2 * 96u + oi) * 96u + oj) * 384u + nn] = acc[a][bq][rr] + bias;
      }
    }
}

// ---------------------------------------------------------------------------
extern "C" void kernel_launch(void* const* d_in, const int* in_sizes, int n_in,
                              void* d_out, int out_size, void* d_ws, size_t ws_size,
                              hipStream_t stream) {
  (void)in_sizes; (void)n_in; (void)out_size; (void)ws_size;
  const float* x        = (const float*)d_in[0];
  const int*   pix_map  = (const int*)  d_in[1];
  const float* w_qkv    = (const float*)d_in[2];
  const float* pos_emb  = (const float*)d_in[3];
  const float* pix_emb  = (const float*)d_in[4];
  const float* w_out    = (const float*)d_in[5];
  const float* b_out    = (const float*)d_in[6];
  float* out = (float*)d_out;

  const size_t QE = (size_t)B_ * HEADS_ * NW_ * 144 * 32;  // 28,311,552 elems
  unsigned short* Qw    = (unsigned short*)d_ws;
  unsigned short* Kw    = Qw + QE;
  unsigned short* Vw    = Kw + QE;
  unsigned short* AO    = Vw + QE;                 // same elem count (73728*384)
  unsigned short* wqkvT = AO + QE;
  unsigned short* woutT = wqkvT + (size_t)NQKV_ * DIM_;

  const int convN = NQKV_ * DIM_ + INNER_ * DIM_;
  hipLaunchKernelGGL(k_convert, dim3((convN + 255) / 256), dim3(256), 0, stream,
                     w_qkv, w_out, wqkvT, woutT);
  hipLaunchKernelGGL(k_qkv, dim3(NQKV_ / 64, MTOT_ / 64), dim3(128), 0, stream,
                     x, wqkvT, Qw, Kw, Vw);
  hipLaunchKernelGGL(k_attn, dim3(B_ * HEADS_ * NW_), dim3(288), 0, stream,
                     Qw, Kw, Vw, pix_map, pos_emb, pix_emb, AO);
  hipLaunchKernelGGL(k_out, dim3(INNER_ / 64, MTOT_ / 64), dim3(128), 0, stream,
                     AO, woutT, b_out, out);
}
